// StrictRetriever_1503238553828
// MI455X (gfx1250) — matrix operand order repaired and checked by reading, not compiled, not run on hardware
//
#include <hip/hip_runtime.h>
#include <hip/hip_bf16.h>
#include <math.h>

// ---- problem sizes (fixed by reference setup_inputs) ----
#define BN      8192      // batch
#define LN      512       // seq len
#define DN      64        // feature dim
#define MN      5000      // memory entries
#define MPAD    5120      // padded so every wave gets exactly 40 of 320 tiles
#define NTILES  (MPAD/16) // 320
#define NPAIRS  20        // 40 tiles per wave, processed 2 per iteration
#define DR      64        // d_repr
#define THRESH  0.95f

typedef __attribute__((ext_vector_type(16))) _Float16 v16h;
typedef __attribute__((ext_vector_type(8)))  _Float16 v8h;
typedef __attribute__((ext_vector_type(8)))  float    v8f;

// ---------------------------------------------------------------------------
// Kernel 1: encode. One block per batch element. Streams 128 KB of query,
// computes xf = mean_D, stats over L, 5->64 projection, LayerNorm, L2 norm,
// writes qn as f16. Pure HBM-bandwidth phase (1.07 GB total read).
// ---------------------------------------------------------------------------
__global__ __launch_bounds__(256) void encode_kernel(
    const float* __restrict__ q,
    const float* __restrict__ proj_w,  // [64,5]
    const float* __restrict__ proj_b,  // [64]
    const float* __restrict__ ln_w,    // [64]
    const float* __restrict__ ln_b,    // [64]
    _Float16* __restrict__ qn16)       // [BN,64]
{
    __shared__ float xf[LN];
    __shared__ float rS[256], rQ[256], rMx[256], rMn[256];
    __shared__ float stats[5];
    __shared__ float hBuf[DR];
    __shared__ float sc[2];
    __shared__ float yBuf[DR];
    __shared__ float nrmInv;

    const int b    = blockIdx.x;
    const int t    = threadIdx.x;
    const int w    = t >> 5;
    const int lane = t & 31;
    const float* qb = q + (size_t)b * (LN * DN);

    // xf[l] = mean over D. Lanes 0-15 / 16-31 of each wave handle two rows;
    // 32 lanes load 512 contiguous bytes per iteration (fully coalesced).
    for (int iter = 0; iter < 32; ++iter) {
        const int pair = iter * 8 + w;             // 0..255
        const int l    = pair * 2 + (lane >> 4);   // 0..511
        const float4* p = (const float4*)(qb + (size_t)l * DN) + (lane & 15);
        float4 v = *p;
        float s = v.x + v.y + v.z + v.w;
        s += __shfl_xor(s, 8);
        s += __shfl_xor(s, 4);
        s += __shfl_xor(s, 2);
        s += __shfl_xor(s, 1);
        if ((lane & 15) == 0) xf[l] = s * (1.0f / 64.0f);
    }
    __syncthreads();

    // block-wide stats over 512 xf values
    {
        const float a = xf[t], c = xf[t + 256];
        rS[t]  = a + c;
        rQ[t]  = a * a + c * c;
        rMx[t] = fmaxf(a, c);
        rMn[t] = fminf(a, c);
    }
    __syncthreads();
    for (int off = 128; off > 0; off >>= 1) {
        if (t < off) {
            rS[t] += rS[t + off];
            rQ[t] += rQ[t + off];
            rMx[t] = fmaxf(rMx[t], rMx[t + off]);
            rMn[t] = fminf(rMn[t], rMn[t + off]);
        }
        __syncthreads();
    }
    if (t == 0) {
        const float mean = rS[0] * (1.0f / (float)LN);
        const float ss   = fmaxf(rQ[0] - (float)LN * mean * mean, 0.0f);
        stats[0] = mean;
        stats[1] = sqrtf(ss / (float)(LN - 1));   // unbiased std
        stats[2] = rMx[0];
        stats[3] = rMn[0];
        stats[4] = xf[LN - 1] - xf[0];            // trend
    }
    __syncthreads();

    // h = stats @ proj_w.T + proj_b
    if (t < DR) {
        float h = proj_b[t];
        #pragma unroll
        for (int s = 0; s < 5; ++s) h += stats[s] * proj_w[t * 5 + s];
        hBuf[t] = h;
    }
    __syncthreads();
    if (t == 0) {
        float mu = 0.0f;
        for (int j = 0; j < DR; ++j) mu += hBuf[j];
        mu *= (1.0f / (float)DR);
        float var = 0.0f;
        for (int j = 0; j < DR; ++j) { const float d = hBuf[j] - mu; var += d * d; }
        var *= (1.0f / (float)DR);
        sc[0] = mu;
        sc[1] = 1.0f / sqrtf(var + 1e-5f);
    }
    __syncthreads();
    if (t < DR) yBuf[t] = ln_w[t] * (hBuf[t] - sc[0]) * sc[1] + ln_b[t];
    __syncthreads();
    if (t == 0) {
        float s = 0.0f;
        for (int j = 0; j < DR; ++j) s += yBuf[j] * yBuf[j];
        nrmInv = 1.0f / fmaxf(sqrtf(s), 1e-12f);
    }
    __syncthreads();
    if (t < DR) qn16[(size_t)b * DR + t] = (_Float16)(yBuf[t] * nrmInv);
}

// ---------------------------------------------------------------------------
// Kernel 2: L2-normalize memory keys -> f16, zero-pad rows [MN, MPAD).
// One wave per row.
// ---------------------------------------------------------------------------
__global__ __launch_bounds__(256) void normkeys_kernel(
    const float* __restrict__ keys,   // [MN,64]
    _Float16* __restrict__ mk16)      // [MPAD,64]
{
    const int w    = threadIdx.x >> 5;
    const int lane = threadIdx.x & 31;
    const int row  = blockIdx.x * 8 + w;      // 0..MPAD-1
    if (row >= MPAD) return;

    float2 v = make_float2(0.0f, 0.0f);
    if (row < MN) v = *((const float2*)(keys + (size_t)row * DN) + lane);
    float s = v.x * v.x + v.y * v.y;
    s += __shfl_xor(s, 16);
    s += __shfl_xor(s, 8);
    s += __shfl_xor(s, 4);
    s += __shfl_xor(s, 2);
    s += __shfl_xor(s, 1);
    const float scale = (row < MN) ? (1.0f / fmaxf(sqrtf(s), 1e-12f)) : 0.0f;
    mk16[(size_t)row * DN + lane * 2 + 0] = (_Float16)(v.x * scale);
    mk16[(size_t)row * DN + lane * 2 + 1] = (_Float16)(v.y * scale);
}

// ---------------------------------------------------------------------------
// Kernel 3a: zero the is_valid flag (stream-ordered before kernel 3b).
// ---------------------------------------------------------------------------
__global__ void init_flag_kernel(float* __restrict__ out)
{
    out[2 * BN * 3] = 0.0f;
}

// ---------------------------------------------------------------------------
// Kernel 3b: fused WMMA GEMM (qn @ keysT) + per-row top-3 + value gather +
// global validity flag. One block = 16 batch rows; 8 waves each own exactly
// 40 of the 320 N-tiles, processed 2 per iteration (two independent
// accumulators -> 4 v_wmma per iteration; tile A's cndmask chain fills
// tile B's WMMA hazard slots and vice versa). Branchless top-3 throughout.
// ---------------------------------------------------------------------------
__global__ __launch_bounds__(256) void sim_topk_kernel(
    const _Float16* __restrict__ qn16,   // [BN,64]
    const _Float16* __restrict__ mk16,   // [MPAD,64]
    const float*    __restrict__ memvals,// [MN,1]
    float*          __restrict__ out)
{
    __shared__ float sV[16][8][16][3];
    __shared__ int   sI[16][8][16][3];

    const int w       = threadIdx.x >> 5;
    const int lane    = threadIdx.x & 31;
    const int rowBase = blockIdx.x * 16;
    const int nlo     = lane & 15;
    const int hi      = lane >> 4;

    // ---- A fragment (contiguous K: lanes 0-15 K=0..15, lanes 16-31 K=16..31), reused for all tiles ----
    const int kb = hi * 16;
    const _Float16* ap = qn16 + (size_t)(rowBase + nlo) * DN;
    const v8h p0 = *(const v8h*)(ap + kb);
    const v8h p1 = *(const v8h*)(ap + 8 + kb);
    const v8h p2 = *(const v8h*)(ap + 32 + kb);
    const v8h p3 = *(const v8h*)(ap + 40 + kb);
    v16h a0, a1;
    #pragma unroll
    for (int i = 0; i < 8; ++i) {
        a0[i] = p0[i]; a0[i + 8] = p1[i];
        a1[i] = p2[i]; a1[i + 8] = p3[i];
    }

    float tv[8][3];
    int   ti[8][3];
    #pragma unroll
    for (int r = 0; r < 8; ++r)
        #pragma unroll
        for (int j = 0; j < 3; ++j) { tv[r][j] = -INFINITY; ti[r][j] = 0; }

    // branchless top-3 insert for one 16x16 C tile (strict > keeps lowest
    // index on ties: within a lane candidates arrive in increasing col order)
    auto insertTile = [&](const v8f& c, int col) {
        const bool valid = (col < MN);
        #pragma unroll
        for (int r = 0; r < 8; ++r) {
            const float v  = valid ? c[r] : -INFINITY;
            const bool g0 = v > tv[r][0];
            const bool g1 = v > tv[r][1];
            const bool g2 = v > tv[r][2];
            const float n2 = g1 ? tv[r][1] : (g2 ? v   : tv[r][2]);
            const int   m2 = g1 ? ti[r][1] : (g2 ? col : ti[r][2]);
            const float n1 = g0 ? tv[r][0] : (g1 ? v   : tv[r][1]);
            const int   m1 = g0 ? ti[r][0] : (g1 ? col : ti[r][1]);
            const float n0 = g0 ? v   : tv[r][0];
            const int   m0 = g0 ? col : ti[r][0];
            tv[r][0] = n0; ti[r][0] = m0;
            tv[r][1] = n1; ti[r][1] = m1;
            tv[r][2] = n2; ti[r][2] = m2;
        }
    };

    // ---- uniform tile loop: every wave does 20 iterations x 2 tiles ----
    const int koff = hi * 16;   // B layout: lanes 0-15 K=0..15, lanes 16-31 K=16..31
    for (int it = 0; it < NPAIRS; ++it) {
        const int tileA = w + it * 16;       // tiles w, w+16, ...
        const int colA  = tileA * 16 + nlo;
        const int colB  = colA + 128;        // tileA + 8
        const _Float16* bpA = mk16 + (size_t)colA * DN + koff;
        const _Float16* bpB = mk16 + (size_t)colB * DN + koff;
        const v16h b0A = *(const v16h*)bpA;
        const v16h b1A = *(const v16h*)(bpA + 32);
        const v16h b0B = *(const v16h*)bpB;
        const v16h b1B = *(const v16h*)(bpB + 32);
        v8f cA = {}, cB = {};
        cA = __builtin_amdgcn_wmma_f32_16x16x32_f16(false, a0, false, b0A,
                                                    (short)0, cA, false, false);
        cA = __builtin_amdgcn_wmma_f32_16x16x32_f16(false, a1, false, b1A,
                                                    (short)0, cA, false, false);
        cB = __builtin_amdgcn_wmma_f32_16x16x32_f16(false, a0, false, b0B,
                                                    (short)0, cB, false, false);
        cB = __builtin_amdgcn_wmma_f32_16x16x32_f16(false, a1, false, b1B,
                                                    (short)0, cB, false, false);
        insertTile(cA, colA);
        insertTile(cB, colB);
    }

    // ---- dump per-lane partial top-3 to LDS ----
    #pragma unroll
    for (int r = 0; r < 8; ++r) {
        const int m = r + hi * 8;
        #pragma unroll
        for (int j = 0; j < 3; ++j) {
            sV[m][w][nlo][j] = tv[r][j];
            sI[m][w][nlo][j] = ti[r][j];
        }
    }
    __syncthreads();

    // ---- merge stage 1: 256 threads, each reduces its (row, lane-chunk)
    //      slice across the 8 waves (24 candidates -> top-3, in place) ----
    {
        const int m  = threadIdx.x >> 4;   // row 0..15
        const int lc = threadIdx.x & 15;   // lane-chunk 0..15
        float v0 = -INFINITY, v1 = -INFINITY, v2 = -INFINITY;
        int   i0 = 0, i1 = 0, i2 = 0;
        #pragma unroll
        for (int ww = 0; ww < 8; ++ww)
            #pragma unroll
            for (int j = 0; j < 3; ++j) {
                const float cv = sV[m][ww][lc][j];
                const int   ci = sI[m][ww][lc][j];
                const bool b0 = (cv > v0) || (cv == v0 && ci < i0);
                const bool b1 = (cv > v1) || (cv == v1 && ci < i1);
                const bool b2 = (cv > v2) || (cv == v2 && ci < i2);
                const float n2 = b1 ? v1 : (b2 ? cv : v2);
                const int   m2 = b1 ? i1 : (b2 ? ci : i2);
                const float n1 = b0 ? v0 : (b1 ? cv : v1);
                const int   m1 = b0 ? i0 : (b1 ? ci : i1);
                const float n0 = b0 ? cv : v0;
                const int   m0 = b0 ? ci : i0;
                v0 = n0; i0 = m0;
                v1 = n1; i1 = m1;
                v2 = n2; i2 = m2;
            }
        // write back into this thread's own slice (wave slot 0): no conflicts
        sV[m][0][lc][0] = v0; sI[m][0][lc][0] = i0;
        sV[m][0][lc][1] = v1; sI[m][0][lc][1] = i1;
        sV[m][0][lc][2] = v2; sI[m][0][lc][2] = i2;
    }
    __syncthreads();

    // ---- merge stage 2: one thread per row, 48 candidates -> top-3 ----
    if (threadIdx.x < 16) {
        const int m = threadIdx.x;
        float v0 = -INFINITY, v1 = -INFINITY, v2 = -INFINITY;
        int   i0 = 0, i1 = 0, i2 = 0;
        for (int lc = 0; lc < 16; ++lc)
            #pragma unroll
            for (int j = 0; j < 3; ++j) {
                const float cv = sV[m][0][lc][j];
                const int   ci = sI[m][0][lc][j];
                const bool b0 = (cv > v0) || (cv == v0 && ci < i0);
                const bool b1 = (cv > v1) || (cv == v1 && ci < i1);
                const bool b2 = (cv > v2) || (cv == v2 && ci < i2);
                const float n2 = b1 ? v1 : (b2 ? cv : v2);
                const int   m2 = b1 ? i1 : (b2 ? ci : i2);
                const float n1 = b0 ? v0 : (b1 ? cv : v1);
                const int   m1 = b0 ? i0 : (b1 ? ci : i1);
                const float n0 = b0 ? cv : v0;
                const int   m0 = b0 ? ci : i0;
                v0 = n0; i0 = m0;
                v1 = n1; i1 = m1;
                v2 = n2; i2 = m2;
            }
        const int grow = rowBase + m;
        // retrieved = memory_values[top_idx]  -> out[0 .. BN*3)
        out[grow * 3 + 0] = memvals[i0];
        out[grow * 3 + 1] = memvals[i1];
        out[grow * 3 + 2] = memvals[i2];
        // top_sim -> out[BN*3 .. 2*BN*3)
        const int off = BN * 3;
        out[off + grow * 3 + 0] = v0;
        out[off + grow * 3 + 1] = v1;
        out[off + grow * 3 + 2] = v2;
        // is_valid flag -> out[2*BN*3] (benign racy 1.0f stores)
        if (v0 > THRESH) out[2 * off] = 1.0f;
    }
}

// ---------------------------------------------------------------------------
extern "C" void kernel_launch(void* const* d_in, const int* in_sizes, int n_in,
                              void* d_out, int out_size, void* d_ws, size_t ws_size,
                              hipStream_t stream) {
    const float* query  = (const float*)d_in[0];
    const float* mkeys  = (const float*)d_in[1];
    const float* mvals  = (const float*)d_in[2];
    const float* proj_w = (const float*)d_in[3];
    const float* proj_b = (const float*)d_in[4];
    const float* ln_w   = (const float*)d_in[5];
    const float* ln_b   = (const float*)d_in[6];
    float* out = (float*)d_out;

    // workspace: qn f16 [8192*64] then keys f16 [5120*64] (keeps 128B align)
    _Float16* qn16 = (_Float16*)d_ws;
    _Float16* mk16 = qn16 + (size_t)BN * DR;

    encode_kernel<<<dim3(BN), dim3(256), 0, stream>>>(query, proj_w, proj_b,
                                                      ln_w, ln_b, qn16);
    normkeys_kernel<<<dim3(MPAD / 8), dim3(256), 0, stream>>>(mkeys, mk16);
    init_flag_kernel<<<dim3(1), dim3(1), 0, stream>>>(out);
    sim_topk_kernel<<<dim3(BN / 16), dim3(256), 0, stream>>>(qn16, mk16, mvals, out);
}
